// eHON_MPL_boundary_6622839570871
// MI455X (gfx1250) — matrix operand order repaired and checked by reading, not compiled, not run on hardware
//
#include <hip/hip_runtime.h>
#include <hip/hip_bf16.h>

typedef __attribute__((ext_vector_type(16))) _Float16 v16h;
typedef __attribute__((ext_vector_type(8)))  float    v8f;

#define NCELL 50000
#define NEDGE 262144
#define FD 128
#define HD 128

// ---------------------------------------------------------------------------
// Prepass: zero message accumulators, seed x_out = x
// ---------------------------------------------------------------------------
__global__ void init_kernel(float* __restrict__ m_up, float* __restrict__ m_dn,
                            const float* __restrict__ x, float* __restrict__ x_out,
                            int n_nodes) {
  int i = blockIdx.x * blockDim.x + threadIdx.x;
  if (i < n_nodes * FD) { m_up[i] = 0.0f; m_dn[i] = 0.0f; }
  if (i < n_nodes * 3)  { x_out[i] = x[i]; }
}

// ---------------------------------------------------------------------------
// Prepass: W [K][128] f32 row-major  ->  Wh [128][K] f16 (transposed, k-contig)
// ---------------------------------------------------------------------------
__global__ void wcvt_kernel(const float* __restrict__ W, _Float16* __restrict__ Wh, int K) {
  int idx = blockIdx.x * blockDim.x + threadIdx.x;
  if (idx < K * 128) {
    int n = idx / K;
    int k = idx % K;
    Wh[n * K + k] = (_Float16)W[k * 128 + n];
  }
}

// ---------------------------------------------------------------------------
// Fragment helpers
// ---------------------------------------------------------------------------
__device__ __forceinline__ void a_half_from_f32(v16h& a, int base, const float* p) {
  float4 u0 = ((const float4*)p)[0];
  float4 u1 = ((const float4*)p)[1];
  a[base + 0] = (_Float16)u0.x; a[base + 1] = (_Float16)u0.y;
  a[base + 2] = (_Float16)u0.z; a[base + 3] = (_Float16)u0.w;
  a[base + 4] = (_Float16)u1.x; a[base + 5] = (_Float16)u1.y;
  a[base + 6] = (_Float16)u1.z; a[base + 7] = (_Float16)u1.w;
}

__device__ __forceinline__ v16h b_from_lds(const _Float16* p) {
  v16h b;
#pragma unroll
  for (int t = 0; t < 16; ++t) b[t] = p[t];
  return b;
}

__device__ __forceinline__ v16h a_from_lds(const _Float16* row, int kb, int hi) {
  v16h a;
  const _Float16* p0 = row + kb + hi * 16;
  const _Float16* p1 = row + kb + hi * 16 + 8;
#pragma unroll
  for (int t = 0; t < 8; ++t) { a[t] = p0[t]; a[8 + t] = p1[t]; }
  return a;
}

#define WMMA(A, B, C) __builtin_amdgcn_wmma_f32_16x16x32_f16(false, (A), false, (B), (short)0, (C), false, false)

// ---------------------------------------------------------------------------
// Fused edge kernel: gather -> msg MLP (WMMA) -> segment-sum m (atomics)
//                    -> coord MLP (WMMA) -> sigmoid gate -> segment-sum dx
// Block = 256 threads (8 waves), 32 edges per wave (2 M-tiles), 256 per block.
// Each B fragment (ds_load_b128 x2) feeds 2 WMMAs; B loads are double-buffered
// so the pending ds_load overlaps WMMA execution.
// ---------------------------------------------------------------------------
__global__ void __launch_bounds__(256)
edge_kernel(const float* __restrict__ h,      // [N,128]
            const float* __restrict__ hn,     // [Nnbr,128]
            const float* __restrict__ x,      // [N,3]
            const float* __restrict__ xn,     // [Nnbr,3]
            const int*   __restrict__ ei,     // [E] segment ids (into N)
            const int*   __restrict__ ej,     // [E] neighbor ids
            const _Float16* __restrict__ W1h, // [128][256]  (transposed, first 256 rows)
            const _Float16* __restrict__ W2h, // [128][128]
            const _Float16* __restrict__ Wc1h,// [128][128]
            const float* __restrict__ W1full, // original [257][128] (for row 256)
            const float* __restrict__ b1,
            const float* __restrict__ b2,
            const float* __restrict__ bc1,
            const float* __restrict__ wc2,    // [128]  (H->1 weight)
            const float* __restrict__ bc2,    // [1]
            const float* __restrict__ coordw, // [2]
            int cw_idx,
            float* __restrict__ m_out,        // [N,128] f32 accumulator
            float* __restrict__ x_out)        // [N,3]   f32 accumulator
{
  extern __shared__ char smem[];
  _Float16* sW1  = (_Float16*)smem;        // 128*256
  _Float16* sW2  = sW1 + 128 * 256;        // 128*128
  _Float16* sWc1 = sW2 + 128 * 128;        // 128*128
  _Float16* sZ   = sWc1 + 128 * 128;       // 8*32*128
  float* sB1     = (float*)(sZ + 8 * 32 * 128); // 128
  float* sB2     = sB1 + 128;
  float* sBc1    = sB2 + 128;
  float* sW1last = sBc1 + 128;             // 128 (row k=256 of W1)
  float* sWc2    = sW1last + 128;          // 128
  float* sDist   = sWc2 + 128;             // 8*32
  float* sDx     = sDist + 8 * 32;         // 8*32*3
  int*   sIdxI   = (int*)(sDx + 8 * 32 * 3); // 8*32
  int*   sIdxJ   = sIdxI + 8 * 32;           // 8*32

  const int tid  = threadIdx.x;
  const int wave = tid >> 5;
  const int lane = tid & 31;
  const int hi   = lane >> 4;
  const int nlo  = lane & 15;

  // ---- cooperative weight staging (vectorized 16B copies) ----
  {
    const uint4* s = (const uint4*)W1h; uint4* d = (uint4*)sW1;
    for (int t = tid; t < 128 * 256 / 8; t += 256) d[t] = s[t];
    s = (const uint4*)W2h; d = (uint4*)sW2;
    for (int t = tid; t < 128 * 128 / 8; t += 256) d[t] = s[t];
    s = (const uint4*)Wc1h; d = (uint4*)sWc1;
    for (int t = tid; t < 128 * 128 / 8; t += 256) d[t] = s[t];
    if (tid < 128) {
      sB1[tid]     = b1[tid];
      sB2[tid]     = b2[tid];
      sBc1[tid]    = bc1[tid];
      sW1last[tid] = W1full[256 * 128 + tid];
      sWc2[tid]    = wc2[tid];
    }
  }

  // ---- per-wave edge meta: indices, dx, ||dx||^2 (one edge per lane) ----
  const int ebase = blockIdx.x * 256 + wave * 32;
  {
    int e = ebase + lane;
    int i = ei[e], j = ej[e];
    sIdxI[wave * 32 + lane] = i;
    sIdxJ[wave * 32 + lane] = j;
    float d0 = x[i * 3 + 0] - xn[j * 3 + 0];
    float d1 = x[i * 3 + 1] - xn[j * 3 + 1];
    float d2 = x[i * 3 + 2] - xn[j * 3 + 2];
    sDx[(wave * 32 + lane) * 3 + 0] = d0;
    sDx[(wave * 32 + lane) * 3 + 1] = d1;
    sDx[(wave * 32 + lane) * 3 + 2] = d2;
    sDist[wave * 32 + lane] = d0 * d0 + d1 * d1 + d2 * d2;
  }
  __syncthreads();

  const float bc2s = bc2[0];
  const float cw   = coordw[cw_idx];

  // A-fragment source rows: tile 0 = rows [0..15], tile 1 = rows [16..31]
  const int m0 = wave * 32 + nlo;
  const int m1 = wave * 32 + 16 + nlo;
  const float* rowh0  = h  + (size_t)sIdxI[m0] * FD;
  const float* rowhn0 = hn + (size_t)sIdxJ[m0] * FD;
  const float* rowh1  = h  + (size_t)sIdxI[m1] * FD;
  const float* rowhn1 = hn + (size_t)sIdxJ[m1] * FD;

  v8f accA[8], accB[8];
#pragma unroll
  for (int nn = 0; nn < 8; ++nn)
#pragma unroll
    for (int r = 0; r < 8; ++r) { accA[nn][r] = 0.0f; accB[nn][r] = 0.0f; }

  // ---- GEMM1: msg[32x256] @ W1[256x128]  (128 WMMAs) ----
#pragma unroll
  for (int kk = 0; kk < 8; ++kk) {
    const int kb = kk * 32;
    const int r0 = kb + hi * 16;
    const int r1 = kb + hi * 16 + 8;
    const float* p00 = (r0 < FD) ? (rowh0 + r0) : (rowhn0 + (r0 - FD));
    const float* p01 = (r1 < FD) ? (rowh0 + r1) : (rowhn0 + (r1 - FD));
    const float* p10 = (r0 < FD) ? (rowh1 + r0) : (rowhn1 + (r0 - FD));
    const float* p11 = (r1 < FD) ? (rowh1 + r1) : (rowhn1 + (r1 - FD));
    v16h a0, a1;
    a_half_from_f32(a0, 0, p00); a_half_from_f32(a0, 8, p01);
    a_half_from_f32(a1, 0, p10); a_half_from_f32(a1, 8, p11);
    v16h b = b_from_lds(&sW1[nlo * 256 + kb + hi * 16]);
#pragma unroll
    for (int nn = 0; nn < 8; ++nn) {
      v16h bn = (nn < 7) ? b_from_lds(&sW1[((nn + 1) * 16 + nlo) * 256 + kb + hi * 16]) : b;
      accA[nn] = WMMA(a0, b, accA[nn]);
      accB[nn] = WMMA(a1, b, accB[nn]);
      b = bn;
    }
  }

  // bias + rank-1 dist2 column + ReLU -> z (f16, LDS)
#pragma unroll
  for (int nn = 0; nn < 8; ++nn) {
    const int col = nn * 16 + nlo;
    const float bb = sB1[col];
    const float wl = sW1last[col];
#pragma unroll
    for (int r = 0; r < 8; ++r) {
      const int MA = r + 8 * hi;
      const int MB = 16 + MA;
      float vA = accA[nn][r] + bb + sDist[wave * 32 + MA] * wl;
      float vB = accB[nn][r] + bb + sDist[wave * 32 + MB] * wl;
      sZ[(wave * 32 + MA) * 128 + col] = (_Float16)fmaxf(vA, 0.0f);
      sZ[(wave * 32 + MB) * 128 + col] = (_Float16)fmaxf(vB, 0.0f);
    }
  }
  __syncthreads();

  const _Float16* zrow0 = &sZ[(wave * 32 + nlo) * 128];
  const _Float16* zrow1 = &sZ[(wave * 32 + 16 + nlo) * 128];

  // ---- GEMM2: z[32x128] @ W2[128x128]  (64 WMMAs) -> atomic segment sum ----
#pragma unroll
  for (int nn = 0; nn < 8; ++nn)
#pragma unroll
    for (int r = 0; r < 8; ++r) { accA[nn][r] = 0.0f; accB[nn][r] = 0.0f; }
#pragma unroll
  for (int kk = 0; kk < 4; ++kk) {
    const int kb = kk * 32;
    v16h a0 = a_from_lds(zrow0, kb, hi);
    v16h a1 = a_from_lds(zrow1, kb, hi);
    v16h b = b_from_lds(&sW2[nlo * 128 + kb + hi * 16]);
#pragma unroll
    for (int nn = 0; nn < 8; ++nn) {
      v16h bn = (nn < 7) ? b_from_lds(&sW2[((nn + 1) * 16 + nlo) * 128 + kb + hi * 16]) : b;
      accA[nn] = WMMA(a0, b, accA[nn]);
      accB[nn] = WMMA(a1, b, accB[nn]);
      b = bn;
    }
  }
#pragma unroll
  for (int nn = 0; nn < 8; ++nn) {
    const int col = nn * 16 + nlo;
    const float bb = sB2[col];
#pragma unroll
    for (int r = 0; r < 8; ++r) {
      const int MA = r + 8 * hi;
      const int MB = 16 + MA;
      atomicAdd(&m_out[(size_t)sIdxI[wave * 32 + MA] * FD + col], accA[nn][r] + bb);
      atomicAdd(&m_out[(size_t)sIdxI[wave * 32 + MB] * FD + col], accB[nn][r] + bb);
    }
  }

  // ---- GEMM3: z @ Wc1 (64 WMMAs) -> ReLU -> dot(wc2) -> sigmoid -> dx sum ----
#pragma unroll
  for (int nn = 0; nn < 8; ++nn)
#pragma unroll
    for (int r = 0; r < 8; ++r) { accA[nn][r] = 0.0f; accB[nn][r] = 0.0f; }
#pragma unroll
  for (int kk = 0; kk < 4; ++kk) {
    const int kb = kk * 32;
    v16h a0 = a_from_lds(zrow0, kb, hi);
    v16h a1 = a_from_lds(zrow1, kb, hi);
    v16h b = b_from_lds(&sWc1[nlo * 128 + kb + hi * 16]);
#pragma unroll
    for (int nn = 0; nn < 8; ++nn) {
      v16h bn = (nn < 7) ? b_from_lds(&sWc1[((nn + 1) * 16 + nlo) * 128 + kb + hi * 16]) : b;
      accA[nn] = WMMA(a0, b, accA[nn]);
      accB[nn] = WMMA(a1, b, accB[nn]);
      b = bn;
    }
  }
  float partA[8], partB[8];
#pragma unroll
  for (int r = 0; r < 8; ++r) { partA[r] = 0.0f; partB[r] = 0.0f; }
#pragma unroll
  for (int nn = 0; nn < 8; ++nn) {
    const int col = nn * 16 + nlo;
    const float bb = sBc1[col];
    const float w2 = sWc2[col];
#pragma unroll
    for (int r = 0; r < 8; ++r) {
      partA[r] += fmaxf(accA[nn][r] + bb, 0.0f) * w2;
      partB[r] += fmaxf(accB[nn][r] + bb, 0.0f) * w2;
    }
  }
#pragma unroll
  for (int r = 0; r < 8; ++r) {
    float pA = partA[r], pB = partB[r];
    pA += __shfl_xor(pA, 1, 32); pA += __shfl_xor(pA, 2, 32);
    pA += __shfl_xor(pA, 4, 32); pA += __shfl_xor(pA, 8, 32);
    pB += __shfl_xor(pB, 1, 32); pB += __shfl_xor(pB, 2, 32);
    pB += __shfl_xor(pB, 4, 32); pB += __shfl_xor(pB, 8, 32);
    if (nlo == 0) {
      const int eA = wave * 32 + r + 8 * hi;
      const int eB = eA + 16;
      const float sA = 1.0f / (1.0f + __expf(-(pA + bc2s)));
      const float sB = 1.0f / (1.0f + __expf(-(pB + bc2s)));
      const int iA = sIdxI[eA], iB = sIdxI[eB];
      atomicAdd(&x_out[iA * 3 + 0], cw * sDx[eA * 3 + 0] * sA);
      atomicAdd(&x_out[iA * 3 + 1], cw * sDx[eA * 3 + 1] * sA);
      atomicAdd(&x_out[iA * 3 + 2], cw * sDx[eA * 3 + 2] * sA);
      atomicAdd(&x_out[iB * 3 + 0], cw * sDx[eB * 3 + 0] * sB);
      atomicAdd(&x_out[iB * 3 + 1], cw * sDx[eB * 3 + 1] * sB);
      atomicAdd(&x_out[iB * 3 + 2], cw * sDx[eB * 3 + 2] * sB);
    }
  }
}

// ---------------------------------------------------------------------------
// Fused cell kernel: h_out = h + MLP(concat[h, m_up, m_dn])
// Block = 256 threads (8 waves), 32 nodes per wave, 256 nodes per block.
// ---------------------------------------------------------------------------
__global__ void __launch_bounds__(256)
cell_kernel(const float* __restrict__ h,      // [N,128]
            const float* __restrict__ m_up,   // [N,128]
            const float* __restrict__ m_dn,   // [N,128]
            const _Float16* __restrict__ W1h, // [128][384]
            const _Float16* __restrict__ W2h, // [128][128]
            const float* __restrict__ b1,
            const float* __restrict__ b2,
            float* __restrict__ h_out,        // [N,128]
            int n_nodes)
{
  extern __shared__ char smem[];
  _Float16* sW1 = (_Float16*)smem;       // 128*384
  _Float16* sW2 = sW1 + 128 * 384;       // 128*128
  _Float16* sZ  = sW2 + 128 * 128;       // 8*32*128
  float* sB1 = (float*)(sZ + 8 * 32 * 128);
  float* sB2 = sB1 + 128;

  const int tid  = threadIdx.x;
  const int wave = tid >> 5;
  const int lane = tid & 31;
  const int hi   = lane >> 4;
  const int nlo  = lane & 15;

  {
    const uint4* s = (const uint4*)W1h; uint4* d = (uint4*)sW1;
    for (int t = tid; t < 128 * 384 / 8; t += 256) d[t] = s[t];
    s = (const uint4*)W2h; d = (uint4*)sW2;
    for (int t = tid; t < 128 * 128 / 8; t += 256) d[t] = s[t];
    if (tid < 128) { sB1[tid] = b1[tid]; sB2[tid] = b2[tid]; }
  }
  __syncthreads();

  const int rowbase = blockIdx.x * 256 + wave * 32;
  int arow0 = rowbase + nlo;       if (arow0 > n_nodes - 1) arow0 = n_nodes - 1;
  int arow1 = rowbase + 16 + nlo;  if (arow1 > n_nodes - 1) arow1 = n_nodes - 1;
  const float* ph0 = h    + (size_t)arow0 * FD;
  const float* pu0 = m_up + (size_t)arow0 * FD;
  const float* pd0 = m_dn + (size_t)arow0 * FD;
  const float* ph1 = h    + (size_t)arow1 * FD;
  const float* pu1 = m_up + (size_t)arow1 * FD;
  const float* pd1 = m_dn + (size_t)arow1 * FD;

  v8f accA[8], accB[8];
#pragma unroll
  for (int nn = 0; nn < 8; ++nn)
#pragma unroll
    for (int r = 0; r < 8; ++r) { accA[nn][r] = 0.0f; accB[nn][r] = 0.0f; }

  // ---- GEMM1: concat[32x384] @ W_c1[384x128]  (192 WMMAs) ----
#pragma unroll
  for (int kk = 0; kk < 12; ++kk) {
    const int kb = kk * 32;
    const int r0 = kb + hi * 16;
    const int r1 = kb + hi * 16 + 8;
    const float* p00 = (r0 < 128) ? (ph0 + r0) : ((r0 < 256) ? (pu0 + r0 - 128) : (pd0 + r0 - 256));
    const float* p01 = (r1 < 128) ? (ph0 + r1) : ((r1 < 256) ? (pu0 + r1 - 128) : (pd0 + r1 - 256));
    const float* p10 = (r0 < 128) ? (ph1 + r0) : ((r0 < 256) ? (pu1 + r0 - 128) : (pd1 + r0 - 256));
    const float* p11 = (r1 < 128) ? (ph1 + r1) : ((r1 < 256) ? (pu1 + r1 - 128) : (pd1 + r1 - 256));
    v16h a0, a1;
    a_half_from_f32(a0, 0, p00); a_half_from_f32(a0, 8, p01);
    a_half_from_f32(a1, 0, p10); a_half_from_f32(a1, 8, p11);
    v16h b = b_from_lds(&sW1[nlo * 384 + kb + hi * 16]);
#pragma unroll
    for (int nn = 0; nn < 8; ++nn) {
      v16h bn = (nn < 7) ? b_from_lds(&sW1[((nn + 1) * 16 + nlo) * 384 + kb + hi * 16]) : b;
      accA[nn] = WMMA(a0, b, accA[nn]);
      accB[nn] = WMMA(a1, b, accB[nn]);
      b = bn;
    }
  }
#pragma unroll
  for (int nn = 0; nn < 8; ++nn) {
    const int col = nn * 16 + nlo;
    const float bb = sB1[col];
#pragma unroll
    for (int r = 0; r < 8; ++r) {
      const int MA = r + 8 * hi;
      const int MB = 16 + MA;
      sZ[(wave * 32 + MA) * 128 + col] = (_Float16)fmaxf(accA[nn][r] + bb, 0.0f);
      sZ[(wave * 32 + MB) * 128 + col] = (_Float16)fmaxf(accB[nn][r] + bb, 0.0f);
    }
  }
  __syncthreads();

  // ---- GEMM2: z[32x128] @ W_c2[128x128]  (64 WMMAs) + residual ----
#pragma unroll
  for (int nn = 0; nn < 8; ++nn)
#pragma unroll
    for (int r = 0; r < 8; ++r) { accA[nn][r] = 0.0f; accB[nn][r] = 0.0f; }
  const _Float16* zrow0 = &sZ[(wave * 32 + nlo) * 128];
  const _Float16* zrow1 = &sZ[(wave * 32 + 16 + nlo) * 128];
#pragma unroll
  for (int kk = 0; kk < 4; ++kk) {
    const int kb = kk * 32;
    v16h a0 = a_from_lds(zrow0, kb, hi);
    v16h a1 = a_from_lds(zrow1, kb, hi);
    v16h b = b_from_lds(&sW2[nlo * 128 + kb + hi * 16]);
#pragma unroll
    for (int nn = 0; nn < 8; ++nn) {
      v16h bn = (nn < 7) ? b_from_lds(&sW2[((nn + 1) * 16 + nlo) * 128 + kb + hi * 16]) : b;
      accA[nn] = WMMA(a0, b, accA[nn]);
      accB[nn] = WMMA(a1, b, accB[nn]);
      b = bn;
    }
  }
#pragma unroll
  for (int nn = 0; nn < 8; ++nn) {
    const int col = nn * 16 + nlo;
    const float bb = sB2[col];
#pragma unroll
    for (int r = 0; r < 8; ++r) {
      const int MA = r + 8 * hi;
      const int MB = 16 + MA;
      const int nodeA = rowbase + MA;
      const int nodeB = rowbase + MB;
      if (nodeA < n_nodes)
        h_out[(size_t)nodeA * FD + col] = h[(size_t)nodeA * FD + col] + accA[nn][r] + bb;
      if (nodeB < n_nodes)
        h_out[(size_t)nodeB * FD + col] = h[(size_t)nodeB * FD + col] + accB[nn][r] + bb;
    }
  }
}

// ---------------------------------------------------------------------------
extern "C" void kernel_launch(void* const* d_in, const int* in_sizes, int n_in,
                              void* d_out, int out_size, void* d_ws, size_t ws_size,
                              hipStream_t stream) {
  const float* h     = (const float*)d_in[0];
  const float* h_up  = (const float*)d_in[1];
  const float* h_dn  = (const float*)d_in[2];
  const float* x     = (const float*)d_in[3];
  const float* x_up  = (const float*)d_in[4];
  const float* x_dn  = (const float*)d_in[5];
  const int*   bui   = (const int*)d_in[6];
  const int*   buj   = (const int*)d_in[7];
  const int*   bdi   = (const int*)d_in[8];
  const int*   bdj   = (const int*)d_in[9];
  const float* Wbu1  = (const float*)d_in[10]; const float* bbu1 = (const float*)d_in[11];
  const float* Wbu2  = (const float*)d_in[12]; const float* bbu2 = (const float*)d_in[13];
  const float* Wbd1  = (const float*)d_in[14]; const float* bbd1 = (const float*)d_in[15];
  const float* Wbd2  = (const float*)d_in[16]; const float* bbd2 = (const float*)d_in[17];
  const float* Wc1   = (const float*)d_in[18]; const float* bc1  = (const float*)d_in[19];
  const float* Wc2   = (const float*)d_in[20]; const float* bc2  = (const float*)d_in[21];
  const float* Wcu1  = (const float*)d_in[22]; const float* bcu1 = (const float*)d_in[23];
  const float* Wcu2  = (const float*)d_in[24]; const float* bcu2 = (const float*)d_in[25];
  const float* Wcd1  = (const float*)d_in[26]; const float* bcd1 = (const float*)d_in[27];
  const float* Wcd2  = (const float*)d_in[28]; const float* bcd2 = (const float*)d_in[29];
  const float* coordw = (const float*)d_in[30];

  const int N = NCELL, E = NEDGE;

  // Workspace layout
  float* m_up = (float*)d_ws;                        // N*128 f32
  float* m_dn = m_up + (size_t)N * FD;               // N*128 f32
  _Float16* Wbu1h = (_Float16*)(m_dn + (size_t)N * FD);
  _Float16* Wbu2h = Wbu1h + 128 * 256;
  _Float16* Wcu1h = Wbu2h + 128 * 128;
  _Float16* Wbd1h = Wcu1h + 128 * 128;
  _Float16* Wbd2h = Wbd1h + 128 * 256;
  _Float16* Wcd1h = Wbd2h + 128 * 128;
  _Float16* Wc1h  = Wcd1h + 128 * 128;
  _Float16* Wc2h  = Wc1h + 128 * 384;

  float* h_out = (float*)d_out;                      // N*128
  float* x_out = h_out + (size_t)N * FD;             // N*3

  // 1) init accumulators + x_out
  init_kernel<<<(N * FD + 255) / 256, 256, 0, stream>>>(m_up, m_dn, x, x_out, N);

  // 2) transpose+convert weights to f16 [n][k]
  wcvt_kernel<<<(256 * 128 + 255) / 256, 256, 0, stream>>>(Wbu1, Wbu1h, 256);
  wcvt_kernel<<<(128 * 128 + 255) / 256, 256, 0, stream>>>(Wbu2, Wbu2h, 128);
  wcvt_kernel<<<(128 * 128 + 255) / 256, 256, 0, stream>>>(Wcu1, Wcu1h, 128);
  wcvt_kernel<<<(256 * 128 + 255) / 256, 256, 0, stream>>>(Wbd1, Wbd1h, 256);
  wcvt_kernel<<<(128 * 128 + 255) / 256, 256, 0, stream>>>(Wbd2, Wbd2h, 128);
  wcvt_kernel<<<(128 * 128 + 255) / 256, 256, 0, stream>>>(Wcd1, Wcd1h, 128);
  wcvt_kernel<<<(384 * 128 + 255) / 256, 256, 0, stream>>>(Wc1, Wc1h, 384);
  wcvt_kernel<<<(128 * 128 + 255) / 256, 256, 0, stream>>>(Wc2, Wc2h, 128);

  // 3) fused edge kernels (up, down): 256 edges per block
  const int EDGE_SMEM =
      (128 * 256 + 128 * 128 + 128 * 128 + 8 * 32 * 128) * 2  // f16 weights + z
      + (5 * 128 + 8 * 32 + 8 * 32 * 3) * 4                   // biases/aux f32
      + (2 * 8 * 32) * 4;                                     // index arrays
  edge_kernel<<<E / 256, 256, EDGE_SMEM, stream>>>(
      h, h_up, x, x_up, bui, buj, Wbu1h, Wbu2h, Wcu1h, Wbu1,
      bbu1, bbu2, bcu1, Wcu2, bcu2, coordw, 0, m_up, x_out);
  edge_kernel<<<E / 256, 256, EDGE_SMEM, stream>>>(
      h, h_dn, x, x_dn, bdi, bdj, Wbd1h, Wbd2h, Wcd1h, Wbd1,
      bbd1, bbd2, bcd1, Wcd2, bcd2, coordw, 1, m_dn, x_out);

  // 4) fused cell kernel (residual MLP): 256 nodes per block
  const int CELL_SMEM =
      (128 * 384 + 128 * 128 + 8 * 32 * 128) * 2 + 2 * 128 * 4;
  cell_kernel<<<(N + 255) / 256, 256, CELL_SMEM, stream>>>(
      h, m_up, m_dn, Wc1h, Wc2h, bc1, bc2, h_out, N);
}
